// ScalarDotProductPolicyNetworkDualAttention_18614388261538
// MI455X (gfx1250) — matrix operand order repaired and checked by reading, not compiled, not run on hardware
//
#include <hip/hip_runtime.h>
#include <hip/hip_bf16.h>
#include <math.h>

// ---------------------------------------------------------------------------
// Dual-attention policy net, one workgroup (256 thr = 8 wave32) per batch b.
// All GEMMs via v_wmma_f32_16x16x32_bf16. Every GEMM except score_o is oriented so its
// epilogue is a TRANSPOSED tile write = one packed 16B ds_store per lane.
// ---------------------------------------------------------------------------

typedef __attribute__((ext_vector_type(8)))  __bf16 v8bf;
typedef __attribute__((ext_vector_type(16))) __bf16 v16bf;
typedef __attribute__((ext_vector_type(8)))  float  v8f;

#define NWAVE 8

// A fragment: 16x32 bf16, row-major source [M][K] with leading dim ld.
static __device__ __forceinline__ v16bf frag_a(const __bf16* base, int ld, int k0, int lane) {
  const int m = lane & 15, h = lane >> 4;
  const __bf16* p = base + m * ld + k0 + h * 8;
  v8bf lo = *(const v8bf*)(p);
  v8bf hi = *(const v8bf*)(p + 16);
  return __builtin_shufflevector(lo, hi, 0,1,2,3,4,5,6,7,8,9,10,11,12,13,14,15);
}

// B fragment: 32x16 bf16 read from a TRANSPOSED image BT[N][K], leading dim ld.
static __device__ __forceinline__ v16bf frag_b(const __bf16* baseT, int ld, int k0, int lane) {
  const int n = lane & 15, h = lane >> 4;
  const __bf16* p = baseT + n * ld + k0 + h * 8;
  v8bf lo = *(const v8bf*)(p);
  v8bf hi = *(const v8bf*)(p + 16);
  return __builtin_shufflevector(lo, hi, 0,1,2,3,4,5,6,7,8,9,10,11,12,13,14,15);
}

// Tiled GEMM: C[Mt*16 x Ntl*16] = A * B with B given transposed (BT[N][K]).
// Epilogue receives (n, mb, acc): column n = nt*16+(lane&15),
// row base mb = mt*16 + 8*(lane>>4); acc[r] is element (mb+r, n).
template <typename Epi>
static __device__ __forceinline__ void gemm_bf16(const __bf16* A, int lda,
                                                 const __bf16* BT, int ldb,
                                                 int Mt, int Ntl, int K,
                                                 int wave, int lane, Epi&& epi) {
  for (int t = wave; t < Mt * Ntl; t += NWAVE) {
    const int mt = t / Ntl, nt = t % Ntl;
    v8f acc = {0.f, 0.f, 0.f, 0.f, 0.f, 0.f, 0.f, 0.f};
    for (int k0 = 0; k0 < K; k0 += 32) {
      v16bf a = frag_a(A + mt * 16 * lda, lda, k0, lane);
      v16bf b = frag_b(BT + nt * 16 * ldb, ldb, k0, lane);
      acc = __builtin_amdgcn_wmma_f32_16x16x32_bf16(false, a, false, b,
                                                    (short)0, acc, false, false);
    }
    epi(nt * 16 + (lane & 15), mt * 16 + ((lane >> 4) << 3), acc);
  }
}

// Load W[rows][cols] (f32, row-major) into LDS as bf16 transposed:
// dst[c*rows + r]  (image [cols][rows], leading dim = rows).
static __device__ __forceinline__ void load_wT(const float* __restrict__ W, __bf16* dst,
                                               int rows, int cols, int tid) {
  for (int i = tid; i < rows * cols; i += 256) {
    const int r = i / cols, c = i - r * cols;
    dst[c * rows + r] = (__bf16)W[i];
  }
}

__global__ __launch_bounds__(256)
void dual_attn_policy_kernel(const float* __restrict__ states,        // [B,64,128]
                             const float* __restrict__ states_other,  // [B,128,128]
                             const float* __restrict__ Wk,  const float* __restrict__ Wq,
                             const float* __restrict__ Wv,  const float* __restrict__ Wko,
                             const float* __restrict__ Wqo, const float* __restrict__ Wvo,
                             const float* __restrict__ Wfc1, const float* __restrict__ Wfc2,
                             float* __restrict__ outP,   // [B,64,16]
                             float* __restrict__ outW,   // [B,64,64]
                             float* __restrict__ outWo)  // [B,64,128]
{
  const int b    = blockIdx.x;
  const int tid  = threadIdx.x;
  const int wave = tid >> 5;
  const int lane = tid & 31;
  const float scale = 0.08838834764831845f;  // 1/sqrt(128)

  extern __shared__ char smem[];
  __bf16* sS   = (__bf16*)(smem);             // states        [64][128]  16KB
  __bf16* sSo  = (__bf16*)(smem + 16384);     // states_other [128][128]  32KB
  __bf16* sW   = (__bf16*)(smem + 49152);     // weight^T (reused)        32KB
  __bf16* sq   = (__bf16*)(smem + 81920);     // q    [i][d]  ld128       16KB
  __bf16* skB  = (__bf16*)(smem + 98304);     // k    [j][d]  ld128       16KB
  __bf16* svT  = (__bf16*)(smem + 114688);    // v^T  [d][j]  ld64        16KB
  __bf16* sqo  = (__bf16*)(smem + 131072);    // q_o  [i][d]  ld128       16KB
  __bf16* skoB = (__bf16*)(smem + 147456);    // k_o  [kk][d] ld128       32KB
  __bf16* svoT = (__bf16*)(smem + 180224);    // v_o^T [d][kk] ld128      32KB
  float*  sF   = (float*)(smem + 212992);     // f32 scratch              32KB
  __bf16* swgt = (__bf16*)(smem + 245760);    // weight bf16 [i][j] ld64   8KB
  __bf16* swgo = (__bf16*)(smem + 253952);    // weight_o [i][kk] ld128   16KB
  __bf16* snode= (__bf16*)(smem + 270336);    // node [i][256] ld256      32KB
  __bf16* sh   = (__bf16*)(smem + 303104);    // h [i][64] ld64            8KB
                                              // total: 311296 B

  const float* st = states       + (size_t)b * 64 * 128;
  const float* so = states_other + (size_t)b * 128 * 128;

  // Warm L2/WGP$ for the (block-shared) weights while we stage activations.
  {
    const int pf = tid * 64;             // 256 thr * 64 f32 = whole 128x128 matrix
    __builtin_prefetch(Wk  + pf, 0, 3);
    __builtin_prefetch(Wv  + pf, 0, 3);
    __builtin_prefetch(Wqo + pf, 0, 3);
    __builtin_prefetch(Wko + pf, 0, 3);
    __builtin_prefetch(Wvo + pf, 0, 3);
    __builtin_prefetch(Wfc1 + pf, 0, 3);
    __builtin_prefetch(Wfc2 + (tid & 15) * 64, 0, 3);
  }

  // ---- stage inputs (f32 -> bf16) + first weight ----
  for (int i = tid; i < 64 * 128;  i += 256) sS[i]  = (__bf16)st[i];
  for (int i = tid; i < 128 * 128; i += 256) sSo[i] = (__bf16)so[i];
  load_wT(Wq, sW, 128, 128, tid);            // sW = Wq^T [d][in]
  __syncthreads();

  // ---- q^T = Wq^T @ S^T : A=sW [d][in], BT=sS [i][in]; write q[i][d] packed ----
  gemm_bf16(sW, 128, sS, 128, 8, 4, 128, wave, lane,
    [&](int n, int mb, v8f acc) {            // n=i, mb=d
      v8bf pk;
      #pragma unroll
      for (int r = 0; r < 8; ++r) pk[r] = (__bf16)acc[r];
      *(v8bf*)(sq + n * 128 + mb) = pk;
    });
  __syncthreads();
  load_wT(Wk, sW, 128, 128, tid);
  __syncthreads();

  // ---- k^T = Wk^T @ S^T ; write k[j][d] packed ----
  gemm_bf16(sW, 128, sS, 128, 8, 4, 128, wave, lane,
    [&](int n, int mb, v8f acc) {            // n=j, mb=d
      v8bf pk;
      #pragma unroll
      for (int r = 0; r < 8; ++r) pk[r] = (__bf16)acc[r];
      *(v8bf*)(skB + n * 128 + mb) = pk;
    });
  __syncthreads();
  load_wT(Wv, sW, 128, 128, tid);
  __syncthreads();

  // ---- v = tanh(S @ Wv) : A=sS [i][in], BT=sW [d][in]; write v^T[d][j] packed ----
  gemm_bf16(sS, 128, sW, 128, 4, 8, 128, wave, lane,
    [&](int n, int mb, v8f acc) {            // n=d, mb=j
      v8bf pk;
      #pragma unroll
      for (int r = 0; r < 8; ++r) pk[r] = (__bf16)tanhf(acc[r]);
      *(v8bf*)(svT + n * 64 + mb) = pk;
    });
  __syncthreads();
  load_wT(Wqo, sW, 128, 128, tid);
  __syncthreads();

  // ---- q_o^T = Wqo^T @ S^T ; write q_o[i][d] packed ----
  gemm_bf16(sW, 128, sS, 128, 8, 4, 128, wave, lane,
    [&](int n, int mb, v8f acc) {
      v8bf pk;
      #pragma unroll
      for (int r = 0; r < 8; ++r) pk[r] = (__bf16)acc[r];
      *(v8bf*)(sqo + n * 128 + mb) = pk;
    });
  __syncthreads();
  load_wT(Wko, sW, 128, 128, tid);
  __syncthreads();

  // ---- k_o^T = Wko^T @ So^T ; write k_o[kk][d] packed ----
  gemm_bf16(sW, 128, sSo, 128, 8, 8, 128, wave, lane,
    [&](int n, int mb, v8f acc) {            // n=kk, mb=d
      v8bf pk;
      #pragma unroll
      for (int r = 0; r < 8; ++r) pk[r] = (__bf16)acc[r];
      *(v8bf*)(skoB + n * 128 + mb) = pk;
    });
  __syncthreads();
  load_wT(Wvo, sW, 128, 128, tid);
  __syncthreads();

  // ---- v_o = tanh(So @ Wvo) ; write v_o^T[d][kk] packed ----
  gemm_bf16(sSo, 128, sW, 128, 8, 8, 128, wave, lane,
    [&](int n, int mb, v8f acc) {            // n=d, mb=kk
      v8bf pk;
      #pragma unroll
      for (int r = 0; r < 8; ++r) pk[r] = (__bf16)tanhf(acc[r]);
      *(v8bf*)(svoT + n * 128 + mb) = pk;
    });
  __syncthreads();

  // ---- score = q @ k^T : A=sq [i][d], BT=skB [j][d]; write TRANSPOSED f32
  //      image sF[j][i] (ld 64) with packed v8f stores ----
  gemm_bf16(sq, 128, skB, 128, 4, 4, 128, wave, lane,
    [&](int n, int mb, v8f acc) {            // n=j, mb=i
      *(v8f*)(sF + n * 64 + mb) = acc;
    });
  __syncthreads();

  // ---- softmax over j of score/sqrt(D); sF is [j][i] (stride-64 per thread,
  //      conflict-free across the 64 row-threads) ----
  if (tid < 64) {
    const int i = tid;
    float mx = -3.4e38f;
    for (int j = 0; j < 64; ++j) mx = fmaxf(mx, sF[j * 64 + i]);
    float sum = 0.f;
    for (int j = 0; j < 64; ++j) {
      const float e = expf((sF[j * 64 + i] - mx) * scale);
      sF[j * 64 + i] = e; sum += e;
    }
    const float inv = 1.f / sum;
    for (int j = 0; j < 64; ++j) {
      const float w = sF[j * 64 + i] * inv;
      swgt[i * 64 + j] = (__bf16)w;          // weight [i][j] (BT for wav^T)
      outW[(size_t)b * 4096 + i * 64 + j] = w;
    }
  }
  __syncthreads();

  // ---- wav^T = v^T @ weight^T : A=svT [d][j], BT=swgt [i][j];
  //      write node[i][0:128] packed ----
  gemm_bf16(svT, 64, swgt, 64, 8, 4, 64, wave, lane,
    [&](int n, int mb, v8f acc) {            // n=i, mb=d
      v8bf pk;
      #pragma unroll
      for (int r = 0; r < 8; ++r) pk[r] = (__bf16)acc[r];
      *(v8bf*)(snode + n * 256 + mb) = pk;
    });
  __syncthreads();

  // ---- score_o = q_o @ k_o^T -> ROW-major sF[i][kk] (axis=-2 softmax is
  //      conflict-free only in this layout) ----
  gemm_bf16(sqo, 128, skoB, 128, 4, 8, 128, wave, lane,
    [&](int n, int mb, v8f acc) {            // n=kk, mb=i
      #pragma unroll
      for (int r = 0; r < 8; ++r) sF[(mb + r) * 128 + n] = acc[r];
    });
  __syncthreads();

  // ---- softmax over i (COLUMNS, axis=-2 quirk) of score_o/sqrt(D) ----
  if (tid < 128) {
    const int c = tid;
    float mx = -3.4e38f;
    for (int i = 0; i < 64; ++i) mx = fmaxf(mx, sF[i * 128 + c]);
    float sum = 0.f;
    for (int i = 0; i < 64; ++i) {
      const float e = expf((sF[i * 128 + c] - mx) * scale);
      sF[i * 128 + c] = e; sum += e;
    }
    const float inv = 1.f / sum;
    for (int i = 0; i < 64; ++i) {
      const float w = sF[i * 128 + c] * inv;
      swgo[i * 128 + c] = (__bf16)w;         // weight_o [i][kk] (BT for wav_o^T)
      outWo[(size_t)b * 8192 + i * 128 + c] = w;
    }
  }
  __syncthreads();

  // ---- wav_o^T = v_o^T @ weight_o^T : A=svoT [d][kk], BT=swgo [i][kk];
  //      write node[i][128:256] packed ----
  gemm_bf16(svoT, 128, swgo, 128, 8, 4, 128, wave, lane,
    [&](int n, int mb, v8f acc) {            // n=i, mb=d
      v8bf pk;
      #pragma unroll
      for (int r = 0; r < 8; ++r) pk[r] = (__bf16)acc[r];
      *(v8bf*)(snode + n * 256 + 128 + mb) = pk;
    });
  __syncthreads();

  // ---- h^T = Wfc1^T @ node^T : A=sW [hid][256], BT=snode [i][256];
  //      leaky-relu, write h[i][hid] packed ----
  load_wT(Wfc1, sW, 256, 64, tid);           // sW = Wfc1^T [64][256]
  __syncthreads();
  gemm_bf16(sW, 256, snode, 256, 4, 4, 256, wave, lane,
    [&](int n, int mb, v8f acc) {            // n=i, mb=hid
      v8bf pk;
      #pragma unroll
      for (int r = 0; r < 8; ++r) {
        const float x = acc[r];
        pk[r] = (__bf16)(x > 0.f ? x : 0.01f * x);
      }
      *(v8bf*)(sh + n * 64 + mb) = pk;
    });
  __syncthreads();

  // ---- logits^T = Wfc2^T @ h^T : A=sW [16][64], BT=sh [i][64];
  //      write logits[i][16] f32 packed ----
  load_wT(Wfc2, sW, 64, 16, tid);            // sW = Wfc2^T [16][64]
  __syncthreads();
  gemm_bf16(sW, 64, sh, 64, 1, 4, 64, wave, lane,
    [&](int n, int mb, v8f acc) {            // n=i, mb=a (0 or 8)
      *(v8f*)(sF + n * 16 + mb) = acc;
    });
  __syncthreads();

  // ---- policy = softmax over 16 actions ----
  if (tid < 64) {
    const int i = tid;
    float mx = -3.4e38f;
    for (int a = 0; a < 16; ++a) mx = fmaxf(mx, sF[i * 16 + a]);
    float sum = 0.f;
    float e[16];
    #pragma unroll
    for (int a = 0; a < 16; ++a) { e[a] = expf(sF[i * 16 + a] - mx); sum += e[a]; }
    const float inv = 1.f / sum;
    #pragma unroll
    for (int a = 0; a < 16; ++a) outP[(size_t)b * 1024 + i * 16 + a] = e[a] * inv;
  }
}

extern "C" void kernel_launch(void* const* d_in, const int* in_sizes, int n_in,
                              void* d_out, int out_size, void* d_ws, size_t ws_size,
                              hipStream_t stream) {
  const float* states = (const float*)d_in[0];
  const float* states_other = (const float*)d_in[1];
  const float* Wk   = (const float*)d_in[2];
  const float* Wq   = (const float*)d_in[3];
  const float* Wv   = (const float*)d_in[4];
  const float* Wko  = (const float*)d_in[5];
  const float* Wqo  = (const float*)d_in[6];
  const float* Wvo  = (const float*)d_in[7];
  const float* Wfc1 = (const float*)d_in[8];
  const float* Wfc2 = (const float*)d_in[9];

  const int Btot = in_sizes[0] / (64 * 128);
  float* out   = (float*)d_out;
  float* outP  = out;                                   // [B,64,16]
  float* outW  = out  + (size_t)Btot * 64 * 16;         // [B,64,64]
  float* outWo = outW + (size_t)Btot * 64 * 64;         // [B,64,128]

  constexpr size_t SMEM = 311296;  // fits in 320KB WGP LDS
  hipFuncSetAttribute((const void*)dual_attn_policy_kernel,
                      hipFuncAttributeMaxDynamicSharedMemorySize, (int)SMEM);
  dual_attn_policy_kernel<<<Btot, 256, SMEM, stream>>>(
      states, states_other, Wk, Wq, Wv, Wko, Wqo, Wvo, Wfc1, Wfc2,
      outP, outW, outWo);
}
